// RXN_CMPD_MPNN_77043123356003
// MI455X (gfx1250) — matrix operand order repaired and checked by reading, not compiled, not run on hardware
//
#include <hip/hip_runtime.h>
#include <hip/hip_bf16.h>
#include <stdint.h>

// ---------------- problem constants (match reference) ----------------
#define N_ATOMS   50000
#define N_BONDS   100000
#define MAX_NB    6
#define ATOM_FDIM 133
#define BOND_FDIM 147
#define HIDDEN    512
#define DEPTH     5
#define N_MOLS    2048
#define KP_BOND   160   // 147 padded to multiple of 32
#define KP_ATOM   672   // 645 padded to multiple of 32 (max Kp -> sizes LDS tile)

typedef __attribute__((ext_vector_type(16))) __bf16 v16bf;
typedef __attribute__((ext_vector_type(8)))  float  v8f;

union FragBF { uint4 u[2]; v16bf v; };

// ---------------- conversion / padding kernels ----------------

// dst[r*Cp + c] = (bf16)src[r*C + c], zero for c >= C
__global__ void cvt_pad_rowmajor(const float* __restrict__ src, __bf16* __restrict__ dst,
                                 int R, int C, int Cp) {
  long i = (long)blockIdx.x * blockDim.x + threadIdx.x;
  long n = (long)R * Cp;
  if (i >= n) return;
  int r = (int)(i / Cp), c = (int)(i % Cp);
  float v = (c < C) ? src[(long)r * C + c] : 0.f;
  dst[i] = (__bf16)v;
}

// dst[n*Kp + k] = (bf16)src[k*N + n], zero for k >= K  (weights, transposed+padded)
__global__ void cvt_transpose_pad(const float* __restrict__ src, __bf16* __restrict__ dst,
                                  int K, int N, int Kp) {
  long i = (long)blockIdx.x * blockDim.x + threadIdx.x;
  long tot = (long)N * Kp;
  if (i >= tot) return;
  int n = (int)(i / Kp), k = (int)(i % Kp);
  float v = (k < K) ? src[(long)k * N + n] : 0.f;
  dst[i] = (__bf16)v;
}

// ---------------- WMMA GEMM: C[M x 512] = A[M x Kp] * Bt^T ----------------
// A: bf16 row-major [M x Kp]; Bt: bf16 [512 x Kp] (transposed weights).
// Block = 256 threads = 8 waves. Block owns one 16-row A tile, staged once
// into LDS; wave w computes the 16x64 output strip at columns [w*64, w*64+64).
// mode 0: out_f32 = acc (inp), out_bf16 = relu(acc)          (stage 1)
// mode 1: out_bf16 = relu(addend + acc)                      (iterations)
// mode 2: out_f32  = relu(acc + bias[col])                   (final atom GEMM)
__global__ __launch_bounds__(256)
void wmma_gemm_n512(const __bf16* __restrict__ A, const __bf16* __restrict__ Bt, int Kp,
                    const float* __restrict__ addend, const float* __restrict__ bias,
                    float* __restrict__ out_f32, __bf16* __restrict__ out_bf16, int mode) {
  __shared__ __align__(16) __bf16 sA[16 * KP_ATOM];   // 21.5 KB max

  const int lane    = threadIdx.x & 31;
  const int wave    = threadIdx.x >> 5;
  const int tileM   = blockIdx.x;                 // 16 rows per block
  const int halfsel = lane >> 4;                  // 0: lanes 0-15, 1: lanes 16-31
  const int l15     = lane & 15;

  // ---- stage the 16 x Kp A tile (contiguous in memory) into LDS ----
  {
    const uint4* src = (const uint4*)(A + (long)tileM * 16 * Kp);
    uint4*       dst = (uint4*)sA;
    const int chunks = (16 * Kp) >> 3;            // uint4 = 8 bf16
    for (int i = threadIdx.x; i < chunks; i += 256) dst[i] = src[i];
  }
  __syncthreads();

  // A fragment addressing (ISA 16-bit A 16x32 layout), now from LDS:
  //   lane<16 : row M=l15, elems 0..7 = K k0+0..7,  elems 8..15 = K k0+16..23
  //   lane>=16: row M=l15, elems 0..7 = K k0+8..15, elems 8..15 = K k0+24..31
  const int abase = l15 * Kp + (halfsel ? 8 : 0);

  // B fragment addressing: row of Bt = output column, K contiguous:
  //   lane<16 : col N, elems 0..7 = K k0+0..7, 8..15 = K k0+16..23 ; lane>=16: elems 0..7 = K k0+8..15, 8..15 = K k0+24..31
  const int col0 = wave * 64;
  long bbase[4];
  #pragma unroll
  for (int j = 0; j < 4; ++j)
    bbase[j] = (long)(col0 + j * 16 + l15) * Kp + (halfsel ? 8 : 0);

  v8f acc[4];
  #pragma unroll
  for (int j = 0; j < 4; ++j) acc[j] = (v8f){0.f,0.f,0.f,0.f,0.f,0.f,0.f,0.f};

  for (int k0 = 0; k0 < Kp; k0 += 32) {
    FragBF a;
    a.u[0] = *(const uint4*)(sA + abase + k0);        // ds_load_b128
    a.u[1] = *(const uint4*)(sA + abase + k0 + 16);
    #pragma unroll
    for (int j = 0; j < 4; ++j) {
      FragBF b;
      b.u[0] = *(const uint4*)(Bt + bbase[j] + k0);
      b.u[1] = *(const uint4*)(Bt + bbase[j] + k0 + 16);
      acc[j] = __builtin_amdgcn_wmma_f32_16x16x32_bf16(false, a.v, false, b.v,
                                                       (short)0, acc[j], false, false);
    }
  }

  // C/D layout: lane<16: elem v -> (M=v, N=l15); lane>=16: elem v -> (M=8+v, N=l15)
  #pragma unroll
  for (int j = 0; j < 4; ++j) {
    const int outcol = col0 + j * 16 + l15;
    #pragma unroll
    for (int v = 0; v < 8; ++v) {
      int  m   = tileM * 16 + (halfsel ? 8 + v : v);
      long idx = (long)m * HIDDEN + outcol;
      float s  = acc[j][v];
      if (mode == 0) {
        out_f32[idx]  = s;
        out_bf16[idx] = (__bf16)fmaxf(s, 0.f);
      } else if (mode == 1) {
        float t = addend[idx] + s;
        out_bf16[idx] = (__bf16)fmaxf(t, 0.f);
      } else {
        float t = s + bias[outcol];
        out_f32[idx] = fmaxf(t, 0.f);
      }
    }
  }
}

// ---------------- message-passing glue kernels ----------------

// amsg[a][h] = sum_k msg[a2b[a][k]][h]   (f32 accumulate from bf16)
__global__ void gather_atom_sum(const __bf16* __restrict__ msg, const int* __restrict__ a2b,
                                float* __restrict__ amsg) {
  long t = (long)blockIdx.x * blockDim.x + threadIdx.x;
  long tot = (long)N_ATOMS * (HIDDEN / 4);
  if (t >= tot) return;
  int atom = (int)(t >> 7);            // HIDDEN/4 = 128 chunks per atom
  int hb   = ((int)t & 127) << 2;
  float s0 = 0.f, s1 = 0.f, s2 = 0.f, s3 = 0.f;
  #pragma unroll
  for (int k = 0; k < MAX_NB; ++k) {
    int b = a2b[atom * MAX_NB + k];
    const __bf16* p = msg + (long)b * HIDDEN + hb;
    s0 += (float)p[0]; s1 += (float)p[1]; s2 += (float)p[2]; s3 += (float)p[3];
  }
  float* q = amsg + (long)atom * HIDDEN + hb;
  q[0] = s0; q[1] = s1; q[2] = s2; q[3] = s3;
}

// tmp[b][h] = (bf16)( amsg[b2a[b]][h] - (f32)msg[b2revb[b]][h] )
__global__ void bond_update(const float* __restrict__ amsg, const __bf16* __restrict__ msg,
                            const int* __restrict__ b2a, const int* __restrict__ b2revb,
                            __bf16* __restrict__ tmp) {
  long t = (long)blockIdx.x * blockDim.x + threadIdx.x;
  long tot = (long)N_BONDS * (HIDDEN / 4);
  if (t >= tot) return;
  int bond = (int)(t >> 7);
  int hb   = ((int)t & 127) << 2;
  int sa   = b2a[bond];
  int rb   = b2revb[bond];
  const float*  pa = amsg + (long)sa * HIDDEN + hb;
  const __bf16* pr = msg  + (long)rb * HIDDEN + hb;
  __bf16* q = tmp + (long)bond * HIDDEN + hb;
  q[0] = (__bf16)(pa[0] - (float)pr[0]);
  q[1] = (__bf16)(pa[1] - (float)pr[1]);
  q[2] = (__bf16)(pa[2] - (float)pr[2]);
  q[3] = (__bf16)(pa[3] - (float)pr[3]);
}

// atom_in[a][c] = c<133 ? f_atoms[a][c] : (c<645 ? amsg[a][c-133] : 0)
__global__ void build_atom_input(const float* __restrict__ f_atoms,
                                 const float* __restrict__ amsg,
                                 __bf16* __restrict__ atom_in) {
  long t = (long)blockIdx.x * blockDim.x + threadIdx.x;
  long tot = (long)N_ATOMS * KP_ATOM;
  if (t >= tot) return;
  int a = (int)(t / KP_ATOM), c = (int)(t % KP_ATOM);
  float v;
  if (c < ATOM_FDIM)                 v = f_atoms[(long)a * ATOM_FDIM + c];
  else if (c < ATOM_FDIM + HIDDEN)   v = amsg[(long)a * HIDDEN + (c - ATOM_FDIM)];
  else                               v = 0.f;
  atom_in[t] = (__bf16)v;
}

// ---------------- segment mean ----------------
__global__ void zero_f32(float* __restrict__ p, long n) {
  long t = (long)blockIdx.x * blockDim.x + threadIdx.x;
  if (t < n) p[t] = 0.f;
}

__global__ void seg_accum(const float* __restrict__ ah, const int* __restrict__ mol_ids,
                          float* __restrict__ sums) {
  long t = (long)blockIdx.x * blockDim.x + threadIdx.x;
  long tot = (long)N_ATOMS * HIDDEN;
  if (t >= tot) return;
  int atom = (int)(t / HIDDEN), h = (int)(t % HIDDEN);
  atomicAdd(&sums[(long)mol_ids[atom] * HIDDEN + h], ah[t]);
}

__global__ void count_accum(const int* __restrict__ mol_ids, float* __restrict__ counts) {
  long t = (long)blockIdx.x * blockDim.x + threadIdx.x;
  if (t < N_ATOMS) atomicAdd(&counts[mol_ids[t]], 1.0f);
}

__global__ void seg_div(float* __restrict__ out, const float* __restrict__ counts) {
  long t = (long)blockIdx.x * blockDim.x + threadIdx.x;
  long tot = (long)N_MOLS * HIDDEN;
  if (t >= tot) return;
  int m = (int)(t / HIDDEN);
  float c = counts[m];
  out[t] = (c > 0.f) ? out[t] / c : 0.f;
}

// ---------------- launch ----------------
static inline int blocks_for(long n, int bs) { return (int)((n + bs - 1) / bs); }

extern "C" void kernel_launch(void* const* d_in, const int* in_sizes, int n_in,
                              void* d_out, int out_size, void* d_ws, size_t ws_size,
                              hipStream_t stream) {
  const float* f_atoms = (const float*)d_in[0];
  const float* f_bonds = (const float*)d_in[1];
  const int*   a2b     = (const int*)  d_in[2];
  const int*   b2a     = (const int*)  d_in[3];
  const int*   b2revb  = (const int*)  d_in[4];
  const int*   mol_ids = (const int*)  d_in[5];
  const float* W_i     = (const float*)d_in[6];
  const float* W_h     = (const float*)d_in[7];
  const float* W_o     = (const float*)d_in[8];
  const float* b_o     = (const float*)d_in[9];
  float* out = (float*)d_out;

  // workspace layout (all offsets multiples of 256 bytes)
  char* ws = (char*)d_ws;
  size_t off = 0;
  float*  inp     = (float*) (ws + off); off += (size_t)N_BONDS * HIDDEN * 4;   // 204.8 MB
  __bf16* msg     = (__bf16*)(ws + off); off += (size_t)N_BONDS * HIDDEN * 2;   // 102.4 MB
  __bf16* tmp     = (__bf16*)(ws + off); off += (size_t)N_BONDS * HIDDEN * 2;   // 102.4 MB (also atom_in)
  float*  amsg    = (float*) (ws + off); off += (size_t)N_ATOMS * HIDDEN * 4;   // 102.4 MB (also atom_hidden)
  __bf16* abonds  = (__bf16*)(ws + off); off += (size_t)N_BONDS * KP_BOND * 2;  //  32.0 MB
  __bf16* Wi_t    = (__bf16*)(ws + off); off += (size_t)HIDDEN * KP_BOND * 2;
  __bf16* Wh_t    = (__bf16*)(ws + off); off += (size_t)HIDDEN * HIDDEN * 2;
  __bf16* Wo_t    = (__bf16*)(ws + off); off += (size_t)HIDDEN * KP_ATOM * 2;
  float*  counts  = (float*) (ws + off); off += (size_t)N_MOLS * 4;
  (void)ws_size; (void)in_sizes; (void)n_in; (void)out_size;

  const int BS = 256;

  // --- convert inputs/weights to bf16 (padded / transposed) ---
  cvt_pad_rowmajor<<<blocks_for((long)N_BONDS * KP_BOND, BS), BS, 0, stream>>>(
      f_bonds, abonds, N_BONDS, BOND_FDIM, KP_BOND);
  cvt_transpose_pad<<<blocks_for((long)HIDDEN * KP_BOND, BS), BS, 0, stream>>>(
      W_i, Wi_t, BOND_FDIM, HIDDEN, KP_BOND);
  cvt_transpose_pad<<<blocks_for((long)HIDDEN * HIDDEN, BS), BS, 0, stream>>>(
      W_h, Wh_t, HIDDEN, HIDDEN, HIDDEN);
  cvt_transpose_pad<<<blocks_for((long)HIDDEN * KP_ATOM, BS), BS, 0, stream>>>(
      W_o, Wo_t, ATOM_FDIM + HIDDEN, HIDDEN, KP_ATOM);

  // --- stage 1: inp = f_bonds @ W_i ; msg = relu(inp) ---
  wmma_gemm_n512<<<dim3(N_BONDS / 16), 256, 0, stream>>>(
      abonds, Wi_t, KP_BOND, nullptr, nullptr, inp, msg, 0);

  // --- message passing iterations ---
  for (int it = 0; it < DEPTH - 1; ++it) {
    gather_atom_sum<<<blocks_for((long)N_ATOMS * (HIDDEN / 4), BS), BS, 0, stream>>>(
        msg, a2b, amsg);
    bond_update<<<blocks_for((long)N_BONDS * (HIDDEN / 4), BS), BS, 0, stream>>>(
        amsg, msg, b2a, b2revb, tmp);
    wmma_gemm_n512<<<dim3(N_BONDS / 16), 256, 0, stream>>>(
        tmp, Wh_t, HIDDEN, inp, nullptr, nullptr, msg, 1);
  }

  // --- final atom stage ---
  gather_atom_sum<<<blocks_for((long)N_ATOMS * (HIDDEN / 4), BS), BS, 0, stream>>>(
      msg, a2b, amsg);
  __bf16* atom_in = tmp;  // reuse (50000*672*2 = 67.2 MB < 102.4 MB)
  build_atom_input<<<blocks_for((long)N_ATOMS * KP_ATOM, BS), BS, 0, stream>>>(
      f_atoms, amsg, atom_in);
  float* atom_hidden = amsg;  // reuse: amsg fully consumed into atom_in
  wmma_gemm_n512<<<dim3(N_ATOMS / 16), 256, 0, stream>>>(
      atom_in, Wo_t, KP_ATOM, nullptr, b_o, atom_hidden, nullptr, 2);

  // --- segment mean per molecule ---
  zero_f32<<<blocks_for((long)N_MOLS * HIDDEN, BS), BS, 0, stream>>>(out, (long)N_MOLS * HIDDEN);
  zero_f32<<<blocks_for((long)N_MOLS, BS), BS, 0, stream>>>(counts, (long)N_MOLS);
  seg_accum<<<blocks_for((long)N_ATOMS * HIDDEN, BS), BS, 0, stream>>>(atom_hidden, mol_ids, out);
  count_accum<<<blocks_for((long)N_ATOMS, BS), BS, 0, stream>>>(mol_ids, counts);
  seg_div<<<blocks_for((long)N_MOLS * HIDDEN, BS), BS, 0, stream>>>(out, counts);
}
